// ContextAwareSpanClassifier_1030792151083
// MI455X (gfx1250) — compile-verified, matrix operand order checked by reading, not run on hardware
//
#include <hip/hip_runtime.h>
#include <hip/hip_bf16.h>

// ---------------------------------------------------------------------------
// ContextAwareSpanClassifier for MI455X (gfx1250, wave32, WMMA)
// B=4, S=2048, H=768, L=9, M=5 (window 11), CTX_RATIO=0.5
// ---------------------------------------------------------------------------

#define Bc 4
#define Sc 2048
#define Hc 768
#define Lc 9
#define Mc 5
#define WINc 11
#define NROWS (Bc * Sc)      // 8192
#define NEG_INF_F (-1e30f)

typedef __attribute__((ext_vector_type(16))) __bf16 v16bf;
typedef __attribute__((ext_vector_type(8)))  float  v8f;

// ---------------------------------------------------------------------------
// Kernel 1: per row  -> scores = x.wa + ba ; base_logits = x.Wc + bc ;
//                       also cast x -> bf16 for the WMMA GEMM.
// ---------------------------------------------------------------------------
__global__ void k_pre(const float* __restrict__ x, const float* __restrict__ Wc,
                      const float* __restrict__ bc, const float* __restrict__ wa,
                      const float* __restrict__ ba,
                      float* __restrict__ scores, float* __restrict__ baseL,
                      __bf16* __restrict__ xb)
{
    const int row = blockIdx.x;
    const int t   = threadIdx.x;
    const float* xr = x + (size_t)row * Hc;

    float v0 = xr[t], v1 = xr[t + 256], v2 = xr[t + 512];
    xb[(size_t)row * Hc + t]       = (__bf16)v0;
    xb[(size_t)row * Hc + t + 256] = (__bf16)v1;
    xb[(size_t)row * Hc + t + 512] = (__bf16)v2;

    __shared__ float red[256];

    red[t] = v0 * wa[t] + v1 * wa[t + 256] + v2 * wa[t + 512];
    __syncthreads();
    for (int o = 128; o > 0; o >>= 1) { if (t < o) red[t] += red[t + o]; __syncthreads(); }
    if (t == 0) scores[row] = red[0] + ba[0];
    __syncthreads();

    float pl[Lc];
#pragma unroll
    for (int l = 0; l < Lc; ++l)
        pl[l] = v0 * Wc[t * Lc + l] + v1 * Wc[(t + 256) * Lc + l] + v2 * Wc[(t + 512) * Lc + l];
#pragma unroll
    for (int l = 0; l < Lc; ++l) {
        red[t] = pl[l];
        __syncthreads();
        for (int o = 128; o > 0; o >>= 1) { if (t < o) red[t] += red[t + o]; __syncthreads(); }
        if (t == 0) baseL[(size_t)row * Lc + l] = red[0] + bc[l];
        __syncthreads();
    }
}

// ---------------------------------------------------------------------------
// Kernel 2: windowed softmax + weighted context, LDS-tiled.
// Each block owns 16 consecutive output rows; the 26-row halo tile (78 KB of
// f32, fits in the 320 KB WGP LDS) is staged once with the CDNA5 async
// global->LDS path (ASYNCcnt), then all 16x768 outputs are produced from LDS.
// ---------------------------------------------------------------------------
#define TS 16
#define TROWS (TS + 2 * Mc)   // 26

__global__ void k_ctx(const float* __restrict__ x, const float* __restrict__ scores,
                      __bf16* __restrict__ ctxb)
{
    __shared__ float tile[TROWS * Hc];       // 26*768*4 = 79872 B
    __shared__ float wsm[TS * WINc];

    const int blk = blockIdx.x;              // 0..511
    const int sb  = (blk * TS) & (Sc - 1);   // tile start s (tiles never straddle batches)
    const int b0  = ((blk * TS) >> 11) * Sc; // b*S
    const int tid = threadIdx.x;

    // ---- async copy of 26 (edge-clamped) rows into LDS -------------------
    const unsigned ldsBase = (unsigned)(size_t)(void*)tile;
    for (int c = tid; c < TROWS * (Hc / 4); c += 256) {   // 16-byte chunks
        const int r    = c / (Hc / 4);
        const int off4 = c % (Hc / 4);
        int srow = sb - Mc + r;
        srow = srow < 0 ? 0 : (srow > Sc - 1 ? Sc - 1 : srow);
        const float* g = x + (size_t)(b0 + srow) * Hc + off4 * 4;
        const unsigned laddr = ldsBase + (unsigned)c * 16u;
        asm volatile("global_load_async_to_lds_b128 %0, %1, off"
                     :: "v"(laddr), "v"(g) : "memory");
    }
    asm volatile("s_wait_asynccnt 0x0" ::: "memory");

    // ---- per-row softmax weights (threads 0..15, one row each) -----------
    if (tid < TS) {
        const int s = sb + tid;
        float w[WINc];
        float mx = NEG_INF_F;
#pragma unroll
        for (int j = 0; j < WINc; ++j) {
            const int p = s - Mc + j;
            const bool valid = (p >= 0) && (p < Sc);
            const int q = p < 0 ? 0 : (p > Sc - 1 ? Sc - 1 : p);
            w[j] = valid ? scores[b0 + q] : NEG_INF_F;
            mx = fmaxf(mx, w[j]);
        }
        float sum = 0.f;
#pragma unroll
        for (int j = 0; j < WINc; ++j) { w[j] = __expf(w[j] - mx); sum += w[j]; }
        const float inv = 1.f / sum;
#pragma unroll
        for (int j = 0; j < WINc; ++j) wsm[tid * WINc + j] = w[j] * inv;
    }
    __syncthreads();

    // ---- weighted context from LDS --------------------------------------
    for (int c = tid; c < TS * Hc; c += 256) {
        const int rr = c / Hc;
        const int h  = c % Hc;
        float acc = 0.f;
#pragma unroll
        for (int j = 0; j < WINc; ++j)
            acc += wsm[rr * WINc + j] * tile[(rr + j) * Hc + h];
        ctxb[(size_t)(b0 + sb + rr) * Hc + h] = (__bf16)acc;
    }
}

// ---------------------------------------------------------------------------
// Kernel 3: cast W1 (2H x H) to bf16
// ---------------------------------------------------------------------------
__global__ void k_castW1(const float* __restrict__ W1, __bf16* __restrict__ w1b, int n)
{
    int i = blockIdx.x * 256 + threadIdx.x;
    if (i < n) w1b[i] = (__bf16)W1[i];
}

// ---------------------------------------------------------------------------
// Kernel 4: WMMA bf16 GEMM  h = xb @ W1[0:768,:] + ctxb @ W1[768:1535,:] + b1
// Block tile M=128 x N=128, 8 waves, 8 WMMA tiles per wave, double-buffered
// LDS slab (one barrier per 32-K step).  All 16 B-fragment ds_load_b128s are
// issued as a batch before the WMMA chain so the scheduler can use partial
// s_wait_dscnt counts; next-slab global loads are issued after the A loads so
// the first WMMA's loadcnt wait does not cover them.
// ---------------------------------------------------------------------------
#define LDST 40    // LDS row stride in bf16 elems (80 B, multiple of 16 B)
#define BMt 128
#define BNt 128
#define KSTEPS (2 * Hc / 32)   // 48

union QU { uint4 q; __bf16 e[8]; };

__global__ void k_gemm(const __bf16* __restrict__ xb, const __bf16* __restrict__ ctxb,
                       const __bf16* __restrict__ w1b, const float* __restrict__ b1,
                       float* __restrict__ hout)
{
    __shared__ __bf16 ldsBT[2][BNt * LDST];   // 2 x 10240 B

    const int tid    = threadIdx.x;
    const int lane   = tid & 31;
    const int wave   = tid >> 5;
    const int blockM = blockIdx.x * BMt;
    const int blockN = blockIdx.y * BNt;
    const int rowBase = blockM + wave * 16;

    const v8f zero = {0.f, 0.f, 0.f, 0.f, 0.f, 0.f, 0.f, 0.f};
    v8f acc[8];
#pragma unroll
    for (int nt = 0; nt < 8; ++nt) acc[nt] = zero;

    const int ks = (lane >> 4) * 8;                       // K sub-offset per half-wave
    const size_t arowOff = (size_t)(rowBase + (lane & 15)) * Hc;

    // cooperative slab mapping: thread -> (k-line, 16-wide n chunk)
    const int kl = tid >> 3;          // 0..31
    const int n0 = (tid & 7) * 16;    // 0..112

    QU u0, u1;
    auto loadSlab = [&](int kk, QU& a, QU& bq) {
        const __bf16* src = w1b + (size_t)(kk + kl) * Hc + blockN + n0;
        a.q  = *reinterpret_cast<const uint4*>(src);
        bq.q = *reinterpret_cast<const uint4*>(src + 8);
    };
    auto storeSlab = [&](int buf, const QU& a, const QU& bq) {
#pragma unroll
        for (int j = 0; j < 8; ++j) ldsBT[buf][(n0 + j) * LDST + kl] = a.e[j];
#pragma unroll
        for (int j = 0; j < 8; ++j) ldsBT[buf][(n0 + 8 + j) * LDST + kl] = bq.e[j];
    };

    // prologue: fill buffer 0
    loadSlab(0, u0, u1);
    storeSlab(0, u0, u1);
    __syncthreads();

    int buf = 0;
    for (int step = 0; step < KSTEPS; ++step) {
        const int kk    = step * 32;              // 0..1535 over concatenated K
        const int phase = (kk >= Hc);
        const __bf16* A = phase ? ctxb : xb;
        const int ka    = phase ? (kk - Hc) : kk;
        const bool haveNext = (step + 1 < KSTEPS);

        // A fragment first: 16-bit A 16x32 layout (lane>>4 selects K sub-block)
        union { uint4 q[2]; v16bf v; } af;
        const uint4* pa = reinterpret_cast<const uint4*>(A + arowOff + ka + ks);
        af.q[0] = pa[0];     // K = ks .. ks+7
        af.q[1] = pa[2];     // K = ks+16 .. ks+23

        // then next-slab global loads (stay in flight across the WMMA chain)
        QU p0, p1;
        if (haveNext) loadSlab(kk + 32, p0, p1);

        if (haveNext) {
            const __bf16* An = (kk + 32 >= Hc) ? ctxb : xb;
            const int kan    = (kk + 32 >= Hc) ? (kk + 32 - Hc) : (kk + 32);
            __builtin_prefetch(An + arowOff + kan + ks, 0, 1);
        }

        // batch all 16 ds_load_b128 B-fragment loads, then run the WMMA chain
        union BF { uint4 q[2]; v16bf v; } bfrag[8];
#pragma unroll
        for (int nt = 0; nt < 8; ++nt) {
            const uint4* pb = reinterpret_cast<const uint4*>(
                    &ldsBT[buf][(nt * 16 + (lane & 15)) * LDST + ks]);
            bfrag[nt].q[0] = pb[0];
            bfrag[nt].q[1] = pb[2];
        }
#pragma unroll
        for (int nt = 0; nt < 8; ++nt)
            acc[nt] = __builtin_amdgcn_wmma_f32_16x16x32_bf16(
                    false, af.v, false, bfrag[nt].v, (short)0, acc[nt], false, false);

        if (haveNext) storeSlab(buf ^ 1, p0, p1); // fill other buffer
        __syncthreads();                          // one barrier per K-step
        buf ^= 1;
    }

    // epilogue: +b1, store f32 h.  C/D layout: lanes 0-15 -> M=vgpr,
    // lanes 16-31 -> M=vgpr+8; N = lane&15 within tile.
    const int rsel = (lane >> 4) << 3;
#pragma unroll
    for (int nt = 0; nt < 8; ++nt) {
        const int col = blockN + nt * 16 + (lane & 15);
        const float bias = b1[col];
#pragma unroll
        for (int i = 0; i < 8; ++i) {
            const int r = rowBase + i + rsel;
            hout[(size_t)r * Hc + col] = acc[nt][i] + bias;
        }
    }
}

// ---------------------------------------------------------------------------
// Kernel 5: LayerNorm + exact GELU + W2 (H x 9) + 0.5/0.5 mix with base.
// ---------------------------------------------------------------------------
__device__ __forceinline__ float gelu_exact(float v) {
    return 0.5f * v * (1.f + erff(v * 0.70710678118654752f));
}

__global__ void k_post(const float* __restrict__ h, const float* __restrict__ gamma,
                       const float* __restrict__ beta, const float* __restrict__ W2,
                       const float* __restrict__ b2, const float* __restrict__ baseL,
                       float* __restrict__ out)
{
    const int row = blockIdx.x;
    const int t   = threadIdx.x;
    const float* hr = h + (size_t)row * Hc;

    float v0 = hr[t], v1 = hr[t + 256], v2 = hr[t + 512];

    __shared__ float red[256];
    red[t] = v0 + v1 + v2;
    __syncthreads();
    for (int o = 128; o > 0; o >>= 1) { if (t < o) red[t] += red[t + o]; __syncthreads(); }
    const float mu = red[0] * (1.f / Hc);
    __syncthreads();

    float d0 = v0 - mu, d1 = v1 - mu, d2 = v2 - mu;
    red[t] = d0 * d0 + d1 * d1 + d2 * d2;
    __syncthreads();
    for (int o = 128; o > 0; o >>= 1) { if (t < o) red[t] += red[t + o]; __syncthreads(); }
    const float var = red[0] * (1.f / Hc);
    __syncthreads();
    const float rstd = rsqrtf(var + 1e-5f);

    float g0 = gelu_exact(d0 * rstd * gamma[t]       + beta[t]);
    float g1 = gelu_exact(d1 * rstd * gamma[t + 256] + beta[t + 256]);
    float g2 = gelu_exact(d2 * rstd * gamma[t + 512] + beta[t + 512]);

    float part[Lc];
#pragma unroll
    for (int l = 0; l < Lc; ++l)
        part[l] = g0 * W2[t * Lc + l] + g1 * W2[(t + 256) * Lc + l] + g2 * W2[(t + 512) * Lc + l];

#pragma unroll
    for (int l = 0; l < Lc; ++l) {
        red[t] = part[l];
        __syncthreads();
        for (int o = 128; o > 0; o >>= 1) { if (t < o) red[t] += red[t + o]; __syncthreads(); }
        if (t == 0) {
            const float ctx = red[0] + b2[l];
            out[(size_t)row * Lc + l] = 0.5f * baseL[(size_t)row * Lc + l] + 0.5f * ctx;
        }
        __syncthreads();
    }
}

// ---------------------------------------------------------------------------
// Launcher
// ---------------------------------------------------------------------------
extern "C" void kernel_launch(void* const* d_in, const int* in_sizes, int n_in,
                              void* d_out, int out_size, void* d_ws, size_t ws_size,
                              hipStream_t stream)
{
    const float* x     = (const float*)d_in[0];   // (4,2048,768)
    const float* Wc    = (const float*)d_in[1];   // (768,9)
    const float* bc    = (const float*)d_in[2];   // (9,)
    const float* wa    = (const float*)d_in[3];   // (768,)
    const float* ba    = (const float*)d_in[4];   // (1,)
    const float* W1    = (const float*)d_in[5];   // (1536,768)
    const float* b1    = (const float*)d_in[6];   // (768,)
    const float* gamma = (const float*)d_in[7];   // (768,)
    const float* beta  = (const float*)d_in[8];   // (768,)
    const float* W2    = (const float*)d_in[9];   // (768,9)
    const float* b2    = (const float*)d_in[10];  // (9,)
    float* out = (float*)d_out;

    // workspace layout (all offsets 4 KiB aligned)
    char* ws = (char*)d_ws;
    float*  scores = (float*) (ws + 0);                       //   32 KB
    float*  baseL  = (float*) (ws + 32768);                   //  288 KB
    __bf16* xb     = (__bf16*)(ws + 327680);                  // 12.0 MB
    __bf16* ctxb   = (__bf16*)(ws + 12910592);                // 12.0 MB
    __bf16* w1b    = (__bf16*)(ws + 25493504);                //  2.3 MB
    float*  hbuf   = (float*) (ws + 27852800);                // 24.0 MB

    k_pre<<<NROWS, 256, 0, stream>>>(x, Wc, bc, wa, ba, scores, baseL, xb);
    k_castW1<<<(2 * Hc * Hc + 255) / 256, 256, 0, stream>>>(W1, w1b, 2 * Hc * Hc);
    k_ctx<<<NROWS / TS, 256, 0, stream>>>(x, scores, ctxb);

    dim3 ggrid(NROWS / BMt, Hc / BNt);   // 64 x 6
    k_gemm<<<ggrid, 256, 0, stream>>>(xb, ctxb, w1b, b1, hbuf);

    k_post<<<NROWS, 256, 0, stream>>>(hbuf, gamma, beta, W2, b2, baseL, out);
}
